// DNRI_Encoder_55198919688340
// MI455X (gfx1250) — matrix operand order repaired and checked by reading, not compiled, not run on hardware
//
#include <hip/hip_runtime.h>

// ---------------------------------------------------------------------------
// DNRI encoder for MI455X (gfx1250, wave32, WMMA bf16, f32 accumulate).
// bf16 activations end-to-end; gates kept f32 for the LSTM nonlinearity.
// GEMM tiles staged with GLOBAL_LOAD_ASYNC_TO_LDS_B128 (ASYNCcnt pipeline).
// ---------------------------------------------------------------------------

typedef unsigned short u16;
typedef __attribute__((ext_vector_type(16))) __bf16 v16bf;
typedef __attribute__((ext_vector_type(8)))  float  v8f;

#define TM  128   // block tile M
#define TNB 128   // block tile N
#define TK  32    // K step (WMMA bf16 K)
#define LDP 40    // padded LDS row stride (halfwords): 80B -> conflict-free gathers

// Problem constants
#define BATCH 128
#define NV 32
#define NE 992                      // V*(V-1)
#define HID 256
#define MEDGE (BATCH * NE)          // 126976 edge tokens
#define MNODE (BATCH * NV)          // 4096 node tokens

__device__ __forceinline__ u16 f2bf(float f) {
  unsigned int u = __float_as_uint(f);
  u += 0x7FFFu + ((u >> 16) & 1u);      // round-to-nearest-even
  return (u16)(u >> 16);
}
__device__ __forceinline__ float bf2f(u16 u) {
  return __uint_as_float((unsigned)u << 16);
}

union FragBF { v16bf v; uint4 q[2]; };

// ---- CDNA5 async global->LDS copy (ASYNCcnt), ISA cdna5 §10 / §15.18 ------
#define AS3 __attribute__((address_space(3)))

__device__ __forceinline__ unsigned ldsOff(const void* p) {
  // addrspace(3) pointers are raw 32-bit LDS offsets
  return (unsigned)(unsigned long long)(AS3 const void*)p;
}
__device__ __forceinline__ void asyncLd16(unsigned lds_off, const void* gptr) {
  asm volatile("global_load_async_to_lds_b128 %0, %1, off"
               :: "v"(lds_off), "v"((unsigned long long)gptr)
               : "memory");
}
__device__ __forceinline__ void waitAsync0() {
  asm volatile("s_wait_asynccnt 0x0" ::: "memory");
}

// ---------------------------------------------------------------------------
// f32 -> bf16 bulk conversion (n8 = count/8)
// ---------------------------------------------------------------------------
__global__ __launch_bounds__(256)
void f32_to_bf16_kernel(const float* __restrict__ in, u16* __restrict__ out, int n8)
{
  const int gid = blockIdx.x * 256 + threadIdx.x;
  if (gid >= n8) return;
  const float4 a = *(const float4*)(in + (size_t)gid * 8);
  const float4 b = *(const float4*)(in + (size_t)gid * 8 + 4);
  uint4 o;
  o.x = (unsigned)f2bf(a.x) | ((unsigned)f2bf(a.y) << 16);
  o.y = (unsigned)f2bf(a.z) | ((unsigned)f2bf(a.w) << 16);
  o.z = (unsigned)f2bf(b.x) | ((unsigned)f2bf(b.y) << 16);
  o.w = (unsigned)f2bf(b.z) | ((unsigned)f2bf(b.w) << 16);
  *(uint4*)(out + (size_t)gid * 8) = o;
}

// ---------------------------------------------------------------------------
// GEMM: C[M,N] = act(A[M,K]*W[N,K]^T + bias)   (bf16 in, bf16 or f32 out)
// DUAL=1: C = act(A0*W0^T + A1*W1^T + bias0 + bias1)   (same K for both)
// Block 128x128, 8 waves (4M x 2N), wave tile 32x64 = 2x4 WMMA accumulators.
// Double-buffered LDS; tiles staged by async global->LDS copies which are in
// flight while the current buffer's WMMAs execute.
// Requires M%128==0, N%128==0, K%32==0.
// ---------------------------------------------------------------------------
template<int ACT, int OUT_BF16, int DUAL>
__global__ __launch_bounds__(256)
void gemm_bf16_kernel(const u16* __restrict__ A0, const u16* __restrict__ W0,
                      const u16* __restrict__ A1, const u16* __restrict__ W1,
                      const float* __restrict__ bias0, const float* __restrict__ bias1,
                      void* __restrict__ Cout, int M, int N, int K)
{
  __shared__ u16 lA[2][TM * LDP];
  __shared__ u16 lB[2][TNB * LDP];

  const int tid  = threadIdx.x;
  const int lane = tid & 31;
  const int wave = tid >> 5;
  const int wm   = (wave & 3) * 32;        // wave M offset in block tile
  const int wn   = (wave >> 2) * 64;       // wave N offset in block tile
  const int blockM = blockIdx.x * TM;
  const int blockN = blockIdx.y * TNB;

  // staging: 512 chunks of 8 bf16 per 128x32 tile; 2 chunks per thread per tile
  const int ldRow0 = tid >> 2;             // rows 0..63
  const int ldRow1 = ldRow0 + 64;          // rows 64..127
  const int ldC8   = (tid & 3) * 8;        // col offset (halfwords)

  // per-thread LDS byte offsets of the 4 async-copy destinations, per buffer
  unsigned dA0[2], dA1[2], dB0[2], dB1[2];
#pragma unroll
  for (int b = 0; b < 2; ++b) {
    dA0[b] = ldsOff(&lA[b][ldRow0 * LDP + ldC8]);
    dA1[b] = ldsOff(&lA[b][ldRow1 * LDP + ldC8]);
    dB0[b] = ldsOff(&lB[b][ldRow0 * LDP + ldC8]);
    dB1[b] = ldsOff(&lB[b][ldRow1 * LDP + ldC8]);
  }

  v8f acc[2][4];
#pragma unroll
  for (int mt = 0; mt < 2; ++mt)
#pragma unroll
    for (int nt = 0; nt < 4; ++nt)
#pragma unroll
      for (int r = 0; r < 8; ++r)
        acc[mt][nt][r] = 0.0f;

  const int KS = K >> 5;
  const int S  = DUAL ? (2 * KS) : KS;

  auto gasync = [&](int s, int buf) {     // issue async copies for step s
    const int  p  = DUAL ? (s >= KS) : 0;
    const u16* A  = p ? A1 : A0;
    const u16* W  = p ? W1 : W0;
    const int  k0 = (p ? (s - KS) : s) * TK;
    asyncLd16(dA0[buf], A + (size_t)(blockM + ldRow0) * K + k0 + ldC8);
    asyncLd16(dA1[buf], A + (size_t)(blockM + ldRow1) * K + k0 + ldC8);
    asyncLd16(dB0[buf], W + (size_t)(blockN + ldRow0) * K + k0 + ldC8);
    asyncLd16(dB1[buf], W + (size_t)(blockN + ldRow1) * K + k0 + ldC8);
  };

  // per-lane fragment gather offsets (ISA 16-bit A 16x32 / B 32x16 layouts)
  const int aoff = (wm + (lane & 15)) * LDP + (lane >> 4) * 8;
  const int boff = (wn + (lane & 15)) * LDP + (lane >> 4) * 8;

  gasync(0, 0);
  waitAsync0();
  __syncthreads();

  for (int s = 0; s < S; ++s) {
    const int buf = s & 1;
    if (s + 1 < S) gasync(s + 1, buf ^ 1);   // copies fly during WMMAs

    FragBF a[2], b[4];
#pragma unroll
    for (int mt = 0; mt < 2; ++mt) {
      a[mt].q[0] = *(const uint4*)(&lA[buf][aoff + mt * 16 * LDP]);
      a[mt].q[1] = *(const uint4*)(&lA[buf][aoff + mt * 16 * LDP + 16]);
    }
#pragma unroll
    for (int nt = 0; nt < 4; ++nt) {
      b[nt].q[0] = *(const uint4*)(&lB[buf][boff + nt * 16 * LDP]);
      b[nt].q[1] = *(const uint4*)(&lB[buf][boff + nt * 16 * LDP + 16]);
    }
#pragma unroll
    for (int mt = 0; mt < 2; ++mt)
#pragma unroll
      for (int nt = 0; nt < 4; ++nt)
        acc[mt][nt] = __builtin_amdgcn_wmma_f32_16x16x32_bf16(
            false, a[mt].v, false, b[nt].v, (short)0, acc[mt][nt], false, false);

    if (s + 1 < S) waitAsync0();             // next buffer fully landed
    __syncthreads();
  }

  // epilogue: bias + activation + store
#pragma unroll
  for (int mt = 0; mt < 2; ++mt)
#pragma unroll
    for (int nt = 0; nt < 4; ++nt) {
      const int col = blockN + wn + nt * 16 + (lane & 15);
      float bv = bias0[col];
      if (DUAL) bv += bias1[col];
      const int rowBase = blockM + wm + mt * 16 + (lane >> 4) * 8;
      if (OUT_BF16) {
        u16* Cp = (u16*)Cout + (size_t)rowBase * N + col;
#pragma unroll
        for (int r = 0; r < 8; ++r) {
          float v = acc[mt][nt][r] + bv;
          if (ACT) v = fmaxf(v, 0.0f);
          Cp[(size_t)r * N] = f2bf(v);
        }
      } else {
        float* Cp = (float*)Cout + (size_t)rowBase * N + col;
#pragma unroll
        for (int r = 0; r < 8; ++r) {
          float v = acc[mt][nt][r] + bv;
          if (ACT) v = fmaxf(v, 0.0f);
          Cp[(size_t)r * N] = v;
        }
      }
    }
}

// ---------------------------------------------------------------------------
// cat2: out[b,e,:] = [x[b,send], x[b,recv]]  (bf16, 512 cols, 8-elt chunks)
// ---------------------------------------------------------------------------
__global__ __launch_bounds__(256)
void concat2_kernel(const u16* __restrict__ x, u16* __restrict__ out)
{
  const int gid = blockIdx.x * 256 + threadIdx.x;   // 64 chunks / token
  const int t = gid >> 6;
  const int c = gid & 63;
  if (t >= MEDGE) return;
  const int b = t / NE, e = t % NE;
  const int s = e / (NV - 1), rr = e % (NV - 1);
  const int rcv = rr + (rr >= s);
  const int node = (c < 32) ? s : rcv;
  const int h8 = c & 31;
  const uint4 v = *(const uint4*)(x + (size_t)(b * NV + node) * HID + h8 * 8);
  *(uint4*)(out + (size_t)t * 512 + c * 8) = v;
}

// ---------------------------------------------------------------------------
// cat3: out[b,e,:] = [x[b,send], x[b,recv], skip[b,e]]  (bf16, 768 cols)
// ---------------------------------------------------------------------------
__global__ __launch_bounds__(256)
void concat3_kernel(const u16* __restrict__ x, const u16* __restrict__ skip,
                    u16* __restrict__ out)
{
  const int gid = blockIdx.x * 256 + threadIdx.x;   // 96 chunks / token
  const int t = gid / 96;
  const int c = gid % 96;
  if (t >= MEDGE) return;
  const int b = t / NE, e = t % NE;
  const int s = e / (NV - 1), rr = e % (NV - 1);
  const int rcv = rr + (rr >= s);
  uint4 v;
  if (c < 64) {
    const int node = (c < 32) ? s : rcv;
    v = *(const uint4*)(x + (size_t)(b * NV + node) * HID + (c & 31) * 8);
  } else {
    v = *(const uint4*)(skip + (size_t)t * HID + (c - 64) * 8);
  }
  *(uint4*)(out + (size_t)t * 768 + c * 8) = v;
}

// ---------------------------------------------------------------------------
// edge->node: out[b,v,:] = (1/31) * sum_{e: recv(e)==v} xe[b,e,:]   (bf16)
// ---------------------------------------------------------------------------
__global__ __launch_bounds__(256)
void e2n_kernel(const u16* __restrict__ xe, u16* __restrict__ out)
{
  const int gid = blockIdx.x * 256 + threadIdx.x;   // 32 chunks / node token
  const int t = gid >> 5;
  const int c = gid & 31;
  if (t >= MNODE) return;
  const int b = t >> 5, v = t & 31;
  float acc[8];
#pragma unroll
  for (int i = 0; i < 8; ++i) acc[i] = 0.0f;
  for (int s = 0; s < NV; ++s) {
    if (s == v) continue;
    const int e = s * (NV - 1) + (v < s ? v : v - 1);
    const uint4 x = *(const uint4*)(xe + (size_t)(b * NE + e) * HID + c * 8);
    acc[0] += bf2f((u16)(x.x & 0xFFFF)); acc[1] += bf2f((u16)(x.x >> 16));
    acc[2] += bf2f((u16)(x.y & 0xFFFF)); acc[3] += bf2f((u16)(x.y >> 16));
    acc[4] += bf2f((u16)(x.z & 0xFFFF)); acc[5] += bf2f((u16)(x.z >> 16));
    acc[6] += bf2f((u16)(x.w & 0xFFFF)); acc[7] += bf2f((u16)(x.w >> 16));
  }
  const float sc = 1.0f / (float)(NV - 1);
  uint4 o;
  o.x = (unsigned)f2bf(acc[0] * sc) | ((unsigned)f2bf(acc[1] * sc) << 16);
  o.y = (unsigned)f2bf(acc[2] * sc) | ((unsigned)f2bf(acc[3] * sc) << 16);
  o.z = (unsigned)f2bf(acc[4] * sc) | ((unsigned)f2bf(acc[5] * sc) << 16);
  o.w = (unsigned)f2bf(acc[6] * sc) | ((unsigned)f2bf(acc[7] * sc) << 16);
  *(uint4*)(out + (size_t)t * HID + c * 8) = o;
}

// ---------------------------------------------------------------------------
// LSTM elementwise (vectorized, 4 elems/thread).
// gates[t,1024]=i|f|g|o (f32). Writes f32 h1/c1 (outputs) + bf16 h1 copy.
// ---------------------------------------------------------------------------
__device__ __forceinline__ float sigf(float x) { return 1.0f / (1.0f + __expf(-x)); }

__global__ __launch_bounds__(256)
void lstm_kernel(const float* __restrict__ gates, const float* __restrict__ c0,
                 float* __restrict__ h1, float* __restrict__ c1,
                 u16* __restrict__ h1b)
{
  const int gid = blockIdx.x * 256 + threadIdx.x;   // 64 quads / token
  if (gid >= MEDGE * 64) return;
  const int t = gid >> 6;
  const int q = gid & 63;
  const float* g = gates + (size_t)t * 1024 + q * 4;
  const float4 gi = *(const float4*)(g);
  const float4 gf = *(const float4*)(g + 256);
  const float4 gg = *(const float4*)(g + 512);
  const float4 go = *(const float4*)(g + 768);
  const float4 cp = *(const float4*)(c0 + (size_t)t * HID + q * 4);
  float4 cn, hn;
  cn.x = sigf(gf.x) * cp.x + sigf(gi.x) * tanhf(gg.x);
  cn.y = sigf(gf.y) * cp.y + sigf(gi.y) * tanhf(gg.y);
  cn.z = sigf(gf.z) * cp.z + sigf(gi.z) * tanhf(gg.z);
  cn.w = sigf(gf.w) * cp.w + sigf(gi.w) * tanhf(gg.w);
  hn.x = sigf(go.x) * tanhf(cn.x);
  hn.y = sigf(go.y) * tanhf(cn.y);
  hn.z = sigf(go.z) * tanhf(cn.z);
  hn.w = sigf(go.w) * tanhf(cn.w);
  *(float4*)(c1 + (size_t)t * HID + q * 4) = cn;
  *(float4*)(h1 + (size_t)t * HID + q * 4) = hn;
  uint2 hb;
  hb.x = (unsigned)f2bf(hn.x) | ((unsigned)f2bf(hn.y) << 16);
  hb.y = (unsigned)f2bf(hn.z) | ((unsigned)f2bf(hn.w) << 16);
  *(uint2*)(h1b + (size_t)t * HID + q * 4) = hb;
}

// ---------------------------------------------------------------------------
// Final prior layer (N=2): out[t,j] = A[t,:].pw3[j,:] + pb3[j]   (A bf16)
// ---------------------------------------------------------------------------
__global__ __launch_bounds__(256)
void prior_out_kernel(const u16* __restrict__ A, const float* __restrict__ W,
                      const float* __restrict__ b, float* __restrict__ out)
{
  const int gid = blockIdx.x * 256 + threadIdx.x;   // over MEDGE*2
  if (gid >= MEDGE * 2) return;
  const int t = gid >> 1, j = gid & 1;
  const u16* a = A + (size_t)t * HID;
  const float* w = W + j * HID;
  float s = 0.0f;
  for (int k = 0; k < HID; k += 8) {
    const uint4 av = *(const uint4*)(a + k);
    const float4 w0 = *(const float4*)(w + k);
    const float4 w1 = *(const float4*)(w + k + 4);
    s += bf2f((u16)(av.x & 0xFFFF)) * w0.x + bf2f((u16)(av.x >> 16)) * w0.y;
    s += bf2f((u16)(av.y & 0xFFFF)) * w0.z + bf2f((u16)(av.y >> 16)) * w0.w;
    s += bf2f((u16)(av.z & 0xFFFF)) * w1.x + bf2f((u16)(av.z >> 16)) * w1.y;
    s += bf2f((u16)(av.w & 0xFFFF)) * w1.z + bf2f((u16)(av.w >> 16)) * w1.w;
  }
  out[gid] = s + b[j];
}

// ---------------------------------------------------------------------------
// Orchestration
// ---------------------------------------------------------------------------
extern "C" void kernel_launch(void* const* d_in, const int* in_sizes, int n_in,
                              void* d_out, int out_size, void* d_ws, size_t ws_size,
                              hipStream_t stream)
{
  const float* inputs = (const float*)d_in[0];
  const float* h0     = (const float*)d_in[1];
  const float* c0     = (const float*)d_in[2];
  const float* wF[16] = {  // f32 weights needing bf16 conversion
    (const float*)d_in[3],  (const float*)d_in[5],  (const float*)d_in[7],   // m1
    (const float*)d_in[9],  (const float*)d_in[11], (const float*)d_in[13],  // m2
    (const float*)d_in[15], (const float*)d_in[17], (const float*)d_in[19],  // m3
    (const float*)d_in[21], (const float*)d_in[23], (const float*)d_in[25],  // m4
    (const float*)d_in[27], (const float*)d_in[28],                          // wih, whh
    (const float*)d_in[31], (const float*)d_in[33]                           // pw1, pw2
  };
  const int wElems[16] = {
    HID * 64, HID * HID, HID * HID,
    HID * 512, HID * HID, HID * HID,
    HID * HID, HID * HID, HID * HID,
    HID * 768, HID * HID, HID * HID,
    4 * HID * HID, 4 * HID * HID,
    HID * HID, HID * HID
  };
  const float* m1b1 = (const float*)d_in[4],  *m1b2 = (const float*)d_in[6],  *m1b3 = (const float*)d_in[8];
  const float* m2b1 = (const float*)d_in[10], *m2b2 = (const float*)d_in[12], *m2b3 = (const float*)d_in[14];
  const float* m3b1 = (const float*)d_in[16], *m3b2 = (const float*)d_in[18], *m3b3 = (const float*)d_in[20];
  const float* m4b1 = (const float*)d_in[22], *m4b2 = (const float*)d_in[24], *m4b3 = (const float*)d_in[26];
  const float* bih = (const float*)d_in[29], *bhh = (const float*)d_in[30];
  const float* pb1 = (const float*)d_in[32], *pb2 = (const float*)d_in[34];
  const float* pw3 = (const float*)d_in[35], *pb3 = (const float*)d_in[36];

  // ---- workspace allocator (256B aligned) ----
  char* cur = (char*)d_ws;
  auto alloc = [&](size_t bytes) -> void* {
    void* p = cur;
    cur += (bytes + 255) & ~(size_t)255;
    return p;
  };
  u16* wB[16];
  for (int i = 0; i < 16; ++i) wB[i] = (u16*)alloc((size_t)wElems[i] * 2);
  u16* xinB  = (u16*)alloc((size_t)MNODE * 64 * 2);
  u16* h0B   = (u16*)alloc((size_t)MEDGE * HID * 2);
  u16* nodeA = (u16*)alloc((size_t)MNODE * HID * 2);
  u16* nodeB = (u16*)alloc((size_t)MNODE * HID * 2);
  u16* bigB  = (u16*)alloc((size_t)MEDGE * 768 * 2);
  u16* bufC  = (u16*)alloc((size_t)MEDGE * HID * 2);
  u16* bufD  = (u16*)alloc((size_t)MEDGE * HID * 2);
  u16* bufE  = (u16*)alloc((size_t)MEDGE * HID * 2);   // x_skip
  u16* bufF  = (u16*)alloc((size_t)MEDGE * HID * 2);   // m4 out
  u16* h1B   = (u16*)alloc((size_t)MEDGE * HID * 2);
  float* gates = (float*)alloc((size_t)MEDGE * 4 * HID * 4);

  float* out   = (float*)d_out;
  float* prior = out;                                  // [MEDGE, 2]
  float* h1    = out + (size_t)MEDGE * 2;              // [MEDGE, HID]
  float* c1    = h1  + (size_t)MEDGE * HID;            // [MEDGE, HID]

  const dim3 blk(256);
  auto conv = [&](const float* src, u16* dst, size_t n) {
    const int n8 = (int)(n / 8);
    hipLaunchKernelGGL(f32_to_bf16_kernel, dim3((n8 + 255) / 256), blk, 0, stream,
                       src, dst, n8);
  };

  // ---- up-front bf16 conversions (weights + inputs + h0) ----
  for (int i = 0; i < 16; ++i) conv(wF[i], wB[i], (size_t)wElems[i]);
  conv(inputs, xinB, (size_t)MNODE * 64);
  conv(h0,     h0B,  (size_t)MEDGE * HID);

#define GEMM(ACT, OBF, DUAL, A0, W0, A1, W1, B0, B1, C, M, N, K)                      \
  hipLaunchKernelGGL((gemm_bf16_kernel<ACT, OBF, DUAL>), dim3((M) / TM, (N) / TNB),   \
                     blk, 0, stream, A0, W0, A1, W1, B0, B1, C, M, N, K)

  // ---- m1 node MLP ----
  GEMM(1, 1, 0, xinB,  wB[0], nullptr, nullptr, m1b1, nullptr, nodeA, MNODE, HID, 64);
  GEMM(1, 1, 0, nodeA, wB[1], nullptr, nullptr, m1b2, nullptr, nodeB, MNODE, HID, HID);
  GEMM(0, 1, 0, nodeB, wB[2], nullptr, nullptr, m1b3, nullptr, nodeA, MNODE, HID, HID);

  // ---- edge concat [send || recv] ----
  hipLaunchKernelGGL(concat2_kernel, dim3((MEDGE * 64) / 256), blk, 0, stream,
                     nodeA, bigB);

  // ---- m2 edge MLP -> x_skip ----
  GEMM(1, 1, 0, bigB, wB[3], nullptr, nullptr, m2b1, nullptr, bufC, MEDGE, HID, 512);
  GEMM(1, 1, 0, bufC, wB[4], nullptr, nullptr, m2b2, nullptr, bufD, MEDGE, HID, HID);
  GEMM(0, 1, 0, bufD, wB[5], nullptr, nullptr, m2b3, nullptr, bufE, MEDGE, HID, HID);

  // ---- edge -> node aggregation ----
  hipLaunchKernelGGL(e2n_kernel, dim3((MNODE * 32) / 256), blk, 0, stream,
                     bufE, nodeA);

  // ---- m3 node MLP ----
  GEMM(1, 1, 0, nodeA, wB[6], nullptr, nullptr, m3b1, nullptr, nodeB, MNODE, HID, HID);
  GEMM(1, 1, 0, nodeB, wB[7], nullptr, nullptr, m3b2, nullptr, nodeA, MNODE, HID, HID);
  GEMM(0, 1, 0, nodeA, wB[8], nullptr, nullptr, m3b3, nullptr, nodeB, MNODE, HID, HID);

  // ---- edge concat [send || recv || skip] ----
  hipLaunchKernelGGL(concat3_kernel, dim3((MEDGE * 96 + 255) / 256), blk, 0, stream,
                     nodeB, bufE, bigB);

  // ---- m4 edge MLP ----
  GEMM(1, 1, 0, bigB, wB[9],  nullptr, nullptr, m4b1, nullptr, bufC, MEDGE, HID, 768);
  GEMM(1, 1, 0, bufC, wB[10], nullptr, nullptr, m4b2, nullptr, bufD, MEDGE, HID, HID);
  GEMM(0, 1, 0, bufD, wB[11], nullptr, nullptr, m4b3, nullptr, bufF, MEDGE, HID, HID);

  // ---- LSTM gates (fused dual GEMM): x4*wih^T + h0*whh^T + bih + bhh ----
  GEMM(0, 0, 1, bufF, wB[12], h0B, wB[13], bih, bhh, gates, MEDGE, 4 * HID, HID);

  // ---- LSTM elementwise -> h1, c1 (f32 outputs) + bf16 h1 ----
  hipLaunchKernelGGL(lstm_kernel, dim3((MEDGE * 64) / 256), blk, 0, stream,
                     gates, c0, h1, c1, h1B);

  // ---- prior MLP ----
  GEMM(1, 1, 0, h1B,  wB[14], nullptr, nullptr, pb1, nullptr, bufC, MEDGE, HID, HID);
  GEMM(1, 1, 0, bufC, wB[15], nullptr, nullptr, pb2, nullptr, bufD, MEDGE, HID, HID);
  hipLaunchKernelGGL(prior_out_kernel, dim3((MEDGE * 2) / 256), blk, 0, stream,
                     bufD, pw3, pb3, prior);

#undef GEMM
}
